// KANLinear_32392643346862
// MI455X (gfx1250) — compile-verified, matrix operand order checked by reading, not run on hardware
//
#include <hip/hip_runtime.h>
#include <hip/hip_bf16.h>

// ---------------------------------------------------------------------------
// KAN Fourier layer as a fused bf16 WMMA GEMM on gfx1250 (wave32).
//   Y[8192,512] = Basis[8192,16384] x W[16384,512]
//   Basis generated on the fly (Chebyshev recurrence: 1 cos + 1 sin per (b,i)).
//   W = coeffs[2,512,512,16] flattened over (i,k): already B^T row-major.
// ---------------------------------------------------------------------------

#define BATCH   8192
#define IN_DIM  512
#define OUT_DIM 512
#define GRIDK   16
#define KPERD   (IN_DIM * GRIDK)        // 8192 contraction elems per d (cos/sin)

typedef __attribute__((ext_vector_type(16))) __bf16        v16bf;
typedef __attribute__((ext_vector_type(8)))  float         v8f;
typedef __attribute__((ext_vector_type(4)))  unsigned int  v4u;

union Frag { v16bf bf; v4u u[2]; };

__device__ __forceinline__ unsigned f2bf(float f) {
    unsigned u = __float_as_uint(f);
    return ((u + 0x7FFFu + ((u >> 16) & 1u)) >> 16) & 0xFFFFu;   // RNE
}
__device__ __forceinline__ unsigned pack2(float a, float b) {
    return f2bf(a) | (f2bf(b) << 16);
}

// ---------------------------------------------------------------------------
// Prep 1: f32 -> bf16 convert of fourier_coeffs (flat layout preserved).
// 8 elements / thread, b128 stores.
// ---------------------------------------------------------------------------
__global__ __launch_bounds__(256) void kan_convert_w(
        const float* __restrict__ src, unsigned short* __restrict__ dst) {
    size_t i = ((size_t)blockIdx.x * 256 + threadIdx.x) * 8;
    float4 a = *(const float4*)(src + i);
    float4 b = *(const float4*)(src + i + 4);
    v4u o;
    o.x = pack2(a.x, a.y);  o.y = pack2(a.z, a.w);
    o.z = pack2(b.x, b.y);  o.w = pack2(b.z, b.w);
    *(v4u*)(dst + i) = o;
}

// ---------------------------------------------------------------------------
// Prep 2: transpose x[8192][512] -> xT[512][8192] (coalesced basis loads).
// ---------------------------------------------------------------------------
__global__ __launch_bounds__(256) void kan_transpose_x(
        const float* __restrict__ x, float* __restrict__ xT) {
    __shared__ float tile[32][33];
    const int lx = threadIdx.x, ly = threadIdx.y;       // (32, 8)
    const int bi = blockIdx.x;                          // IN_DIM / 32
    const int bb = blockIdx.y;                          // BATCH  / 32
    #pragma unroll
    for (int r = 0; r < 32; r += 8)
        tile[ly + r][lx] = x[(size_t)(bb * 32 + ly + r) * IN_DIM + bi * 32 + lx];
    __syncthreads();
    #pragma unroll
    for (int r = 0; r < 32; r += 8)
        xT[(size_t)(bi * 32 + ly + r) * BATCH + bb * 32 + lx] = tile[lx][ly + r];
}

// ---------------------------------------------------------------------------
// Main fused kernel. Block = 256 threads (8 waves), tile = 128(M) x 128(N).
// Wave grid 4x2: each wave -> 32(M) x 64(N) = 2x4 WMMA frags, f32 accum.
// K loop: 256 steps x (2 features * 16 harmonics) per d.
// ---------------------------------------------------------------------------
#define ROWB 80                       // LDS row stride (bytes): 16B aligned, bank-spread
__global__ __launch_bounds__(256, 1) void kan_wmma_kernel(
        const float*  __restrict__ xT,   // [512][8192]
        const __bf16* __restrict__ W,    // [2][512][8192] bf16
        float*        __restrict__ y) {  // [8192][512]
    __shared__ __align__(16) unsigned char ldsA[2 * 128 * ROWB];  // cos plane, sin plane

    const int t     = threadIdx.x;
    const int lane  = t & 31;
    const int wave  = t >> 5;
    const int lhalf = (lane >> 4) & 1;            // lane-group (K-half select)
    const int l15   = lane & 15;
    const int bn    = blockIdx.x;                 // N block (0..3)
    const int bm    = blockIdx.y;                 // M block (0..63)
    const int waveM = wave >> 1;                  // 0..3  (32 rows each)
    const int waveN = wave & 1;                   // 0..1  (64 cols each)

    // ---- accumulators -----------------------------------------------------
    v8f acc[2][4];
    #pragma unroll
    for (int mf = 0; mf < 2; ++mf)
        #pragma unroll
        for (int nf = 0; nf < 4; ++nf)
            #pragma unroll
            for (int r = 0; r < 8; ++r) acc[mf][nf][r] = 0.f;

    // ---- basis producer mapping ------------------------------------------
    const int b_local = t & 127;                  // which of the 128 batch rows
    const int i_local = t >> 7;                   // which of the 2 features this step
    const float* xcol = xT + (size_t)(bm * 128 + b_local);
    unsigned char* rowC = ldsA + b_local * ROWB + i_local * 32;
    unsigned char* rowS = rowC + 128 * ROWB;

    // ---- A-fragment LDS read base (per WMMA 16-bit A layout) -------------
    const unsigned char* fragBase[2];
    #pragma unroll
    for (int mf = 0; mf < 2; ++mf)
        fragBase[mf] = ldsA + (waveM * 32 + mf * 16 + l15) * ROWB + lhalf * 16;

    // ---- B row pointers (B^T rows are contiguous in W) -------------------
    const __bf16* browC[4];
    const __bf16* browS[4];
    #pragma unroll
    for (int nf = 0; nf < 4; ++nf) {
        const int j = bn * 128 + waveN * 64 + nf * 16 + l15;
        browC[nf] = W + (size_t)j * KPERD;
        browS[nf] = W + (size_t)(OUT_DIM + j) * KPERD;
    }

    // ======================= K loop (256 steps) ===========================
    for (int ib = 0; ib < IN_DIM / 2; ++ib) {
        __syncthreads();   // previous step's frag loads done before overwrite

        // -- generate 16 cos + 16 sin harmonics via angle-addition ---------
        float xv = xcol[(size_t)(2 * ib + i_local) * BATCH];
        float c1 = __cosf(xv);
        float s1 = __sinf(xv);
        float tc = 2.f * c1;
        float cv[16], sv[16];
        cv[0] = c1;            sv[0] = s1;
        cv[1] = tc * c1 - 1.f; sv[1] = tc * s1;
        #pragma unroll
        for (int k = 2; k < 16; ++k) {
            cv[k] = __builtin_fmaf(tc, cv[k - 1], -cv[k - 2]);
            sv[k] = __builtin_fmaf(tc, sv[k - 1], -sv[k - 2]);
        }
        v4u pc0, pc1, ps0, ps1;
        pc0.x = pack2(cv[0], cv[1]);   pc0.y = pack2(cv[2], cv[3]);
        pc0.z = pack2(cv[4], cv[5]);   pc0.w = pack2(cv[6], cv[7]);
        pc1.x = pack2(cv[8], cv[9]);   pc1.y = pack2(cv[10], cv[11]);
        pc1.z = pack2(cv[12], cv[13]); pc1.w = pack2(cv[14], cv[15]);
        ps0.x = pack2(sv[0], sv[1]);   ps0.y = pack2(sv[2], sv[3]);
        ps0.z = pack2(sv[4], sv[5]);   ps0.w = pack2(sv[6], sv[7]);
        ps1.x = pack2(sv[8], sv[9]);   ps1.y = pack2(sv[10], sv[11]);
        ps1.z = pack2(sv[12], sv[13]); ps1.w = pack2(sv[14], sv[15]);
        *(v4u*)(rowC)      = pc0;  *(v4u*)(rowC + 16) = pc1;   // ds_store_b128
        *(v4u*)(rowS)      = ps0;  *(v4u*)(rowS + 16) = ps1;

        __syncthreads();

        // -- A fragments from LDS (two b128 per frag, swizzled K layout) ---
        Frag Ac[2], As[2];
        #pragma unroll
        for (int mf = 0; mf < 2; ++mf) {
            const unsigned char* p = fragBase[mf];
            Ac[mf].u[0] = *(const v4u*)(p);
            Ac[mf].u[1] = *(const v4u*)(p + 32);
            As[mf].u[0] = *(const v4u*)(p + 128 * ROWB);
            As[mf].u[1] = *(const v4u*)(p + 128 * ROWB + 32);
        }

        // -- B fragments straight from global (rows contiguous) ------------
        const int kkoff = ib * 32 + lhalf * 8;         // elements within B^T row
        Frag Bc[4], Bs[4];
        #pragma unroll
        for (int nf = 0; nf < 4; ++nf) {
            const __bf16* pc = browC[nf] + kkoff;
            const __bf16* ps = browS[nf] + kkoff;
            Bc[nf].u[0] = *(const v4u*)(pc);
            Bc[nf].u[1] = *(const v4u*)(pc + 16);
            Bs[nf].u[0] = *(const v4u*)(ps);
            Bs[nf].u[1] = *(const v4u*)(ps + 16);
            __builtin_prefetch((const void*)(pc + 32), 0, 0);  // next K slice
            __builtin_prefetch((const void*)(ps + 32), 0, 0);
        }

        // -- 16 WMMAs: 2 M-frags x 4 N-frags x {cos, sin} -------------------
        #pragma unroll
        for (int mf = 0; mf < 2; ++mf)
            #pragma unroll
            for (int nf = 0; nf < 4; ++nf) {
                acc[mf][nf] = __builtin_amdgcn_wmma_f32_16x16x32_bf16(
                    false, Ac[mf].bf, false, Bc[nf].bf,
                    (short)0, acc[mf][nf], false, false);
                acc[mf][nf] = __builtin_amdgcn_wmma_f32_16x16x32_bf16(
                    false, As[mf].bf, false, Bs[nf].bf,
                    (short)0, acc[mf][nf], false, false);
            }
    }

    // ---- epilogue: D layout -> lane {N=l15, half}, VGPR r -> M ------------
    const int row0 = bm * 128 + waveM * 32 + lhalf * 8;
    const int col0 = bn * 128 + waveN * 64 + l15;
    #pragma unroll
    for (int mf = 0; mf < 2; ++mf)
        #pragma unroll
        for (int nf = 0; nf < 4; ++nf)
            #pragma unroll
            for (int r = 0; r < 8; ++r)
                y[(size_t)(row0 + mf * 16 + r) * OUT_DIM + col0 + nf * 16] =
                    acc[mf][nf][r];
}

// ---------------------------------------------------------------------------
extern "C" void kernel_launch(void* const* d_in, const int* in_sizes, int n_in,
                              void* d_out, int out_size, void* d_ws, size_t ws_size,
                              hipStream_t stream) {
    (void)in_sizes; (void)n_in; (void)out_size; (void)ws_size;
    const float* x      = (const float*)d_in[0];   // [8192, 512]
    const float* coeffs = (const float*)d_in[1];   // [2, 512, 512, 16]
    float*       y      = (float*)d_out;           // [8192, 512]

    // workspace: [0,16MB) bf16 coeffs, [16MB,32MB) xT
    unsigned short* Wbf = (unsigned short*)d_ws;
    float* xT = (float*)((char*)d_ws + (size_t)2 * OUT_DIM * KPERD * sizeof(unsigned short));

    // total coeff elems = 2*512*8192 = 8,388,608 ; 8 per thread, 256/block
    kan_convert_w<<<dim3(4096), dim3(256), 0, stream>>>(coeffs, Wbf);
    kan_transpose_x<<<dim3(IN_DIM / 32, BATCH / 32), dim3(32, 8), 0, stream>>>(x, xT);
    kan_wmma_kernel<<<dim3(OUT_DIM / 128, BATCH / 128), dim3(256), 0, stream>>>(
        xT, (const __bf16*)Wbf, y);
}
